// Block_4294967296263
// MI455X (gfx1250) — matrix operand order repaired and checked by reading, not compiled, not run on hardware
//
#include <hip/hip_runtime.h>
#include <math.h>

// ---------------------------------------------------------------------------
// MI455X (gfx1250). wave32. WMMA bf16 16x16x32 GEMMs with async global->LDS
// staging of B operands (double-buffered, ASYNCcnt-tracked).
// Workspace layout (assumes ws_size >= ~132 MB):
//   [0,   64MB)  : x_nhwc (f32)  -- reused as hh (bf16, 32768x1024) after conv
//   [64,  96MB)  : t_bf16 (32768 x 512 bf16) normalized tokens
//   [96,  98MB)  : expert weights bf16 (2 x 1024 x 512)
//   [98, 130MB)  : folded pointwise weights bf16 (32 x 512 x 1024)
//   [130MB ...)  : Nx, Gx2, scale, sidx, beta, psum
// ---------------------------------------------------------------------------

typedef __bf16 bf16_t;
typedef bf16_t v16bf __attribute__((ext_vector_type(16)));
typedef float  v8f   __attribute__((ext_vector_type(8)));
typedef int    v4i_t __attribute__((__vector_size__(16)));

#define AS1 __attribute__((address_space(1)))
#define AS3 __attribute__((address_space(3)))

#define N_TOK 32768   // B*H*W
#define C_DIM 512
#define EXPD  1024

// ---- async global -> LDS (16B per lane), ASYNCcnt-tracked -----------------
static __device__ __forceinline__ void async_ld_b128(const void* g, void* l) {
#if __has_builtin(__builtin_amdgcn_global_load_async_to_lds_b128)
  __builtin_amdgcn_global_load_async_to_lds_b128(
      (AS1 v4i_t*)g, (AS3 v4i_t*)l, 0, 0);
#else
  uint32_t lds_off = (uint32_t)(uintptr_t)l;   // LDS aperture: low 32b = offset
  asm volatile("global_load_async_to_lds_b128 %0, %1, off"
               :: "v"(lds_off), "v"((uint64_t)(uintptr_t)g) : "memory");
#endif
}

static __device__ __forceinline__ void wait_async0() {
#if __has_builtin(__builtin_amdgcn_s_wait_asynccnt)
  __builtin_amdgcn_s_wait_asynccnt(0);
#else
  asm volatile("s_wait_asynccnt 0" ::: "memory");
#endif
}

// ---- WMMA fragment loaders ------------------------------------------------
// A (MxK, row-major over K): lane<16: K {0..7}U{16..23}; lane>=16: +8
static __device__ __forceinline__ v16bf load_a_frag(const bf16_t* __restrict__ base,
                                                    int ld, int row0, int k0) {
  int lane = threadIdx.x & 31;
  int l = lane & 15, hi = lane >> 4;
  const bf16_t* p = base + (size_t)(row0 + l) * ld + k0 + hi * 8;
  union { v16bf v; uint4 u[2]; } u;
  u.u[0] = *(const uint4*)(p);
  u.u[1] = *(const uint4*)(p + 16);
  return u.v;
}

// B fragment from an LDS tile laid out as [col][32 K values] (64B per col):
// lane holds column (l), lanes 0-15 K {0..7}U{16..23}, lanes 16-31 +8
static __device__ __forceinline__ v16bf load_b_frag_lds(const bf16_t* tile, int col0) {
  int lane = threadIdx.x & 31;
  int l = lane & 15, hi = lane >> 4;
  const bf16_t* p = tile + (size_t)(col0 + l) * 32 + hi * 8;
  union { v16bf v; uint4 u[2]; } u;
  u.u[0] = *(const uint4*)(p);
  u.u[1] = *(const uint4*)(p + 16);
  return u.v;
}

static __device__ __forceinline__ v8f wmma_bf16(v16bf a, v16bf b, v8f c) {
  return __builtin_amdgcn_wmma_f32_16x16x32_bf16(false, a, false, b, (short)0, c,
                                                 false, false);
}

// ---- 256-thread block reduction (8 waves of 32) ---------------------------
static __device__ __forceinline__ float block_reduce_sum256(float v) {
  __shared__ float red[8];
#pragma unroll
  for (int o = 16; o > 0; o >>= 1) v += __shfl_down(v, o, 32);
  int lane = threadIdx.x & 31, wv = threadIdx.x >> 5;
  __syncthreads();
  if (lane == 0) red[wv] = v;
  __syncthreads();
  if (wv == 0) {
    float t = (lane < 8) ? red[lane] : 0.f;
#pragma unroll
    for (int o = 4; o > 0; o >>= 1) t += __shfl_down(t, o, 32);
    if (lane == 0) red[0] = t;
  }
  __syncthreads();
  float r = red[0];
  __syncthreads();
  return r;
}

// ---- K1: NCHW -> NHWC transpose of x --------------------------------------
__global__ void k_transpose(const float* __restrict__ x, float* __restrict__ xn) {
  size_t stride = (size_t)gridDim.x * blockDim.x;
  for (size_t idx = (size_t)blockIdx.x * blockDim.x + threadIdx.x;
       idx < (size_t)N_TOK * C_DIM; idx += stride) {
    int c = (int)(idx & 511);
    size_t t = idx >> 9;
    int b = (int)(t >> 10), hw = (int)(t & 1023);
    xn[idx] = x[((size_t)b * 512 + c) * 1024 + hw];
  }
}

// ---- K1b: expert weights fp32 -> bf16 -------------------------------------
__global__ void k_cvt_w(const float* __restrict__ e0w, const float* __restrict__ e1w,
                        bf16_t* __restrict__ wb) {
  size_t stride = (size_t)gridDim.x * blockDim.x;
  const size_t half = (size_t)EXPD * C_DIM;
  for (size_t idx = (size_t)blockIdx.x * blockDim.x + threadIdx.x;
       idx < 2 * half; idx += stride) {
    float v = (idx < half) ? e0w[idx] : e1w[idx - half];
    wb[idx] = (bf16_t)v;
  }
}

// ---- K2: fused depthwise conv 7x7 + LayerNorm + noisy top-1 gate ----------
__global__ void __launch_bounds__(256)
k_conv_ln_gate(const float* __restrict__ xn, const float* __restrict__ dww,
               const float* __restrict__ dwb, const float* __restrict__ lng,
               const float* __restrict__ lnb, const float* __restrict__ gw,
               const float* __restrict__ gb, const float* __restrict__ noise,
               bf16_t* __restrict__ tb, float* __restrict__ scale,
               int* __restrict__ sidx, float* __restrict__ psum) {
  int t = blockIdx.x;
  int b = t >> 10, hw = t & 1023, h = hw >> 5, w = hw & 31;
  float y[2];
#pragma unroll
  for (int r = 0; r < 2; ++r) {
    int c = threadIdx.x + r * 256;
    float acc = dwb[c];
#pragma unroll
    for (int i = 0; i < 7; ++i) {
      int hh2 = h + i - 3;
      if (hh2 < 0 || hh2 > 31) continue;
#pragma unroll
      for (int j = 0; j < 7; ++j) {
        int ww2 = w + j - 3;
        if (ww2 < 0 || ww2 > 31) continue;
        acc += xn[((size_t)(b * 1024 + hh2 * 32 + ww2)) * 512 + c] *
               dww[c * 49 + i * 7 + j];
      }
    }
    y[r] = acc;
  }
  float s  = block_reduce_sum256(y[0] + y[1]);
  float s2 = block_reduce_sum256(y[0] * y[0] + y[1] * y[1]);
  float mu = s * (1.f / 512.f);
  float var = s2 * (1.f / 512.f) - mu * mu;
  float rstd = rsqrtf(var + 1e-5f);
  float g0 = 0.f, g1 = 0.f;
#pragma unroll
  for (int r = 0; r < 2; ++r) {
    int c = threadIdx.x + r * 256;
    float tn = (y[r] - mu) * rstd * lng[c] + lnb[c];
    tb[(size_t)t * 512 + c] = (bf16_t)tn;
    g0 += tn * gw[c];
    g1 += tn * gw[512 + c];
  }
  g0 = block_reduce_sum256(g0);
  g1 = block_reduce_sum256(g1);
  if (threadIdx.x == 0) {
    float l0 = g0 + gb[0] + noise[2 * t + 0] * 0.1f;
    float l1 = g1 + gb[1] + noise[2 * t + 1] * 0.1f;
    float m = fmaxf(l0, l1);
    float e0 = __expf(l0 - m), e1 = __expf(l1 - m);
    float p0 = e0 / (e0 + e1), p1 = 1.f - p0;
    atomicAdd(psum + 0, p0);
    atomicAdd(psum + 1, p1);
    int idx = (p0 >= p1) ? 0 : 1;     // jax top_k tie -> first index
    scale[t] = idx ? p1 : p0;
    sidx[t] = idx;
  }
}

// ---- K3: expert GEMMs, async-LDS staged B, select, QuickGELU, Gx^2 --------
// Block: 8 waves, one 64-col N-group shared, each wave owns 2 M-tiles (32 rows).
// Grid: (2048/16 m-groups) x (16 n-groups) = 2048 blocks.
__global__ void __launch_bounds__(256)
k_experts(const bf16_t* __restrict__ tb, const bf16_t* __restrict__ wb,
          const float* __restrict__ e0b, const float* __restrict__ e1b,
          const float* __restrict__ scale, const int* __restrict__ sidx,
          bf16_t* __restrict__ hh, float* __restrict__ gx2) {
  __shared__ bf16_t sB[2][2][64][32];   // [buf][expert][col][K] = 16KB
  const int n0   = (blockIdx.x & 15) * 64;
  const int mtg  = blockIdx.x >> 4;
  const int wave = threadIdx.x >> 5;
  const int row0 = mtg * 256 + wave * 32;      // this wave: rows row0..row0+31
  const int tid  = threadIdx.x;

  // stage(kstep, buf): copy 8KB (2 experts x 64 cols x 32 K bf16) async
  auto stage = [&](int k0, int buf) {
#pragma unroll
    for (int r = 0; r < 2; ++r) {
      int t = tid + r * 256;                 // 512 x 16B tasks
      int boff = t * 16;
      int ec = boff >> 6;                    // (expert*64 + col)
      int inner = boff & 63;
      int e = ec >> 6, c = ec & 63;
      const char* g = (const char*)wb + (size_t)e * 1048576 +
                      (size_t)(n0 + c) * 1024 + (size_t)k0 * 2 + inner;
      async_ld_b128(g, (char*)&sB[buf][0][0][0] + boff);
    }
  };

  v8f acc[2][2][4];                          // [mtile][expert][ntile]
#pragma unroll
  for (int m = 0; m < 2; ++m)
#pragma unroll
    for (int e = 0; e < 2; ++e)
#pragma unroll
      for (int j = 0; j < 4; ++j) acc[m][e][j] = {};

  stage(0, 0);
  for (int ks = 0; ks < 16; ++ks) {
    int k0 = ks * 32;
    wait_async0();
    __syncthreads();
    if (ks + 1 < 16) stage((ks + 1) * 32, (ks + 1) & 1);
    int buf = ks & 1;
    v16bf a0 = load_a_frag(tb, 512, row0, k0);
    v16bf a1 = load_a_frag(tb, 512, row0 + 16, k0);
#pragma unroll
    for (int e = 0; e < 2; ++e)
#pragma unroll
      for (int j = 0; j < 4; ++j) {
        v16bf bf = load_b_frag_lds(&sB[buf][e][0][0], j * 16);
        acc[0][e][j] = wmma_bf16(a0, bf, acc[0][e][j]);
        acc[1][e][j] = wmma_bf16(a1, bf, acc[1][e][j]);
      }
  }

  const int lane = threadIdx.x & 31;
  const int l = lane & 15, hi = lane >> 4;
  const int bimg = row0 >> 10;
#pragma unroll
  for (int m = 0; m < 2; ++m) {
    int r0 = row0 + m * 16;
#pragma unroll
    for (int j = 0; j < 4; ++j) {
      int col = n0 + j * 16 + l;
      float b0 = e0b[col], b1 = e1b[col];
      float sq = 0.f;
#pragma unroll
      for (int i = 0; i < 8; ++i) {
        int row = r0 + i + hi * 8;
        int e = sidx[row];
        float p = scale[row];
        float v = p * ((e ? acc[m][1][j][i] : acc[m][0][j][i]) + (e ? b1 : b0));
        float g = v / (1.f + __expf(-1.702f * v));   // QuickGELU
        hh[(size_t)row * 1024 + col] = (bf16_t)g;
        sq += g * g;
      }
      sq += __shfl_xor(sq, 16, 32);
      if (hi == 0) atomicAdd(&gx2[bimg * 1024 + col], sq);
    }
  }
}

// ---- K4a: Nx = Gx / (mean(Gx) + 1e-6), per batch --------------------------
__global__ void __launch_bounds__(256)
k_grn_nx(const float* __restrict__ gx2, float* __restrict__ nx) {
  int b = blockIdx.x;
  float g[4];
  float s = 0.f;
#pragma unroll
  for (int r = 0; r < 4; ++r) {
    int c = threadIdx.x + r * 256;
    g[r] = sqrtf(gx2[b * 1024 + c]);
    s += g[r];
  }
  s = block_reduce_sum256(s);
  float inv = 1.f / (s * (1.f / 1024.f) + 1e-6f);
#pragma unroll
  for (int r = 0; r < 4; ++r) {
    int c = threadIdx.x + r * 256;
    nx[b * 1024 + c] = g[r] * inv;
  }
}

// ---- K4b: beta[d] = pw_b[d] + sum_c grn_b[c]*pw_w[d,c] --------------------
__global__ void __launch_bounds__(256)
k_beta(const float* __restrict__ pww, const float* __restrict__ pwb,
       const float* __restrict__ grnb, float* __restrict__ beta) {
  int d = blockIdx.x;
  float s = 0.f;
  for (int c = threadIdx.x; c < 2048; c += 256)
    s += grnb[c] * pww[(size_t)d * 2048 + c];
  s = block_reduce_sum256(s);
  if (threadIdx.x == 0) beta[d] = pwb[d] + s;
}

// ---- K4c: fold GRN + channel-duplication into per-batch weight ------------
__global__ void k_fold(const float* __restrict__ pww, const float* __restrict__ grng,
                       const float* __restrict__ nx, bf16_t* __restrict__ wf) {
  size_t stride = (size_t)gridDim.x * blockDim.x;
  for (size_t idx = (size_t)blockIdx.x * blockDim.x + threadIdx.x;
       idx < (size_t)32 * 512 * 1024; idx += stride) {
    int c = (int)(idx & 1023);
    int d = (int)((idx >> 10) & 511);
    int b = (int)(idx >> 19);
    float n = nx[b * 1024 + c];
    float v = (1.f + grng[c] * n)        * pww[(size_t)d * 2048 + c] +
              (1.f + grng[c + 1024] * n) * pww[(size_t)d * 2048 + c + 1024];
    wf[idx] = (bf16_t)v;
  }
}

// ---- K5: per-batch pointwise GEMM 1024->512 + beta + residual (NCHW) ------
// Block: 8 waves share one 64-col N-group; each wave owns 2 M-tiles.
// Grid: (2048/16 m-groups) x (8 n-groups) = 1024 blocks.
__global__ void __launch_bounds__(256)
k_pointwise(const bf16_t* __restrict__ hh, const bf16_t* __restrict__ wf,
            const float* __restrict__ beta, const float* __restrict__ x,
            float* __restrict__ out) {
  __shared__ bf16_t sB[2][64][32];             // 8KB
  const int n0   = (blockIdx.x & 7) * 64;
  const int mtg  = blockIdx.x >> 3;
  const int wave = threadIdx.x >> 5;
  const int row0 = mtg * 256 + wave * 32;
  const int bimg = (mtg * 256) >> 10;
  const bf16_t* wfb = wf + (size_t)bimg * 512 * 1024;
  const int tid = threadIdx.x;

  auto stage = [&](int k0, int buf) {          // 4KB: 256 x 16B tasks
    int boff = tid * 16;
    int c = boff >> 6;
    int inner = boff & 63;
    const char* g = (const char*)wfb + (size_t)(n0 + c) * 2048 +
                    (size_t)k0 * 2 + inner;
    async_ld_b128(g, (char*)&sB[buf][0][0] + boff);
  };

  v8f acc[2][4];
#pragma unroll
  for (int m = 0; m < 2; ++m)
#pragma unroll
    for (int j = 0; j < 4; ++j) acc[m][j] = {};

  stage(0, 0);
  for (int ks = 0; ks < 32; ++ks) {
    int k0 = ks * 32;
    wait_async0();
    __syncthreads();
    if (ks + 1 < 32) stage((ks + 1) * 32, (ks + 1) & 1);
    int buf = ks & 1;
    v16bf a0 = load_a_frag(hh, 1024, row0, k0);
    v16bf a1 = load_a_frag(hh, 1024, row0 + 16, k0);
#pragma unroll
    for (int j = 0; j < 4; ++j) {
      v16bf bf = load_b_frag_lds(&sB[buf][0][0], j * 16);
      acc[0][j] = wmma_bf16(a0, bf, acc[0][j]);
      acc[1][j] = wmma_bf16(a1, bf, acc[1][j]);
    }
  }

  const int lane = threadIdx.x & 31;
  const int l = lane & 15, hi = lane >> 4;
#pragma unroll
  for (int m = 0; m < 2; ++m) {
    int r0 = row0 + m * 16;
#pragma unroll
    for (int j = 0; j < 4; ++j) {
      int d = n0 + j * 16 + l;
      float bd = beta[d];
#pragma unroll
      for (int i = 0; i < 8; ++i) {
        int row = r0 + i + hi * 8;
        int hw = row & 1023;
        size_t oidx = ((size_t)bimg * 512 + d) * 1024 + hw;
        out[oidx] = x[oidx] + acc[m][j][i] + bd;
      }
    }
  }
}

// ---- K6: load-balance loss ------------------------------------------------
__global__ void k_loss(const float* __restrict__ psum, float* __restrict__ out) {
  float m0 = psum[0] * (1.f / 32768.f) - 0.5f;
  float m1 = psum[1] * (1.f / 32768.f) - 0.5f;
  out[0] = m0 * m0 + m1 * m1;
}

// ---------------------------------------------------------------------------
extern "C" void kernel_launch(void* const* d_in, const int* in_sizes, int n_in,
                              void* d_out, int out_size, void* d_ws, size_t ws_size,
                              hipStream_t stream) {
  const float* x     = (const float*)d_in[0];
  const float* noise = (const float*)d_in[1];
  const float* dww   = (const float*)d_in[2];
  const float* dwb   = (const float*)d_in[3];
  const float* lng   = (const float*)d_in[4];
  const float* lnb   = (const float*)d_in[5];
  const float* gw    = (const float*)d_in[6];
  const float* gb    = (const float*)d_in[7];
  const float* e0w   = (const float*)d_in[8];
  const float* e0b   = (const float*)d_in[9];
  const float* e1w   = (const float*)d_in[10];
  const float* e1b   = (const float*)d_in[11];
  const float* grng  = (const float*)d_in[12];
  const float* grnb  = (const float*)d_in[13];
  const float* pww   = (const float*)d_in[14];
  const float* pwb   = (const float*)d_in[15];

  char* ws = (char*)d_ws;
  float*  xn    = (float*)(ws);                               // 64MB
  bf16_t* hh    = (bf16_t*)(ws);                              // aliases xn (after K2)
  bf16_t* tb    = (bf16_t*)(ws + ((size_t)64 << 20));         // 32MB
  bf16_t* ewb   = (bf16_t*)(ws + ((size_t)96 << 20));         // 2MB
  bf16_t* wfold = (bf16_t*)(ws + ((size_t)98 << 20));         // 32MB
  float*  nx    = (float*)(ws + ((size_t)130 << 20));
  float*  gx2   = (float*)(ws + ((size_t)130 << 20) + (128 << 10));
  float*  scale = (float*)(ws + ((size_t)130 << 20) + (256 << 10));
  int*    sidx  = (int*)  (ws + ((size_t)130 << 20) + (384 << 10));
  float*  beta  = (float*)(ws + ((size_t)130 << 20) + (512 << 10));
  float*  psum  = (float*)(ws + ((size_t)130 << 20) + (516 << 10));

  (void)hipMemsetAsync(gx2, 0, 32 * 1024 * sizeof(float), stream);
  (void)hipMemsetAsync(psum, 0, 2 * sizeof(float), stream);

  k_transpose   <<<8192, 256, 0, stream>>>(x, xn);
  k_cvt_w       <<<2048, 256, 0, stream>>>(e0w, e1w, ewb);
  k_conv_ln_gate<<<N_TOK, 256, 0, stream>>>(xn, dww, dwb, lng, lnb, gw, gb, noise,
                                            tb, scale, sidx, psum);
  k_experts     <<<2048, 256, 0, stream>>>(tb, ewb, e0b, e1b, scale, sidx, hh, gx2);
  k_grn_nx      <<<32, 256, 0, stream>>>(gx2, nx);
  k_beta        <<<512, 256, 0, stream>>>(pww, pwb, grnb, beta);
  k_fold        <<<8192, 256, 0, stream>>>(pww, grng, nx, wfold);
  k_pointwise   <<<1024, 256, 0, stream>>>(hh, wfold, beta, x, (float*)d_out);
  k_loss        <<<1, 1, 0, stream>>>(psum, (float*)d_out + 16777216);
}
